// CAF_39109972197428
// MI455X (gfx1250) — matrix operand order repaired and checked by reading, not compiled, not run on hardware
//
#include <hip/hip_runtime.h>

#define HWC 6400   // 80*80
#define CCH 256
#define REDC 32

typedef __attribute__((ext_vector_type(16))) __bf16 v16bf;
typedef __attribute__((ext_vector_type(8)))  __bf16 v8bf;
typedef __attribute__((ext_vector_type(8)))  float  v8f;

constexpr float SCALE_L2E = 0.17677669529663687f * 1.4426950408889634f; // RED^-0.5 * log2(e)
constexpr float L2E = 1.4426950408889634f;

// ---------------- workspace layout (bytes, all 256B aligned) ----------------
constexpr size_t SZ_PE  = 32UL * HWC * 4;           // pe fp32 (32 x 6400)
constexpr size_t SZ_QT  = (size_t)HWC * REDC * 2;   // q/k transposed bf16 [n][32]
constexpr size_t SZ_VB  = (size_t)CCH * HWC * 2;    // V bf16 [c][n]
constexpr size_t SZ_VF  = (size_t)CCH * HWC * 4;    // V fp32 [c][n]
constexpr size_t SZ_ST  = (size_t)HWC * 4;          // per-row stats

constexpr size_t OFF_PE     = 0;
constexpr size_t OFF_QT_RGB = OFF_PE     + SZ_PE;
constexpr size_t OFF_KT_RGB = OFF_QT_RGB + SZ_QT;
constexpr size_t OFF_QT_CHM = OFF_KT_RGB + SZ_QT;
constexpr size_t OFF_KT_CHM = OFF_QT_CHM + SZ_QT;
constexpr size_t OFF_VB_RGB = OFF_KT_CHM + SZ_QT;
constexpr size_t OFF_VB_CHM = OFF_VB_RGB + SZ_VB;
constexpr size_t OFF_VF_RGB = OFF_VB_CHM + SZ_VB;
constexpr size_t OFF_VF_CHM = OFF_VF_RGB + SZ_VF;
constexpr size_t OFF_MX_RGB = OFF_VF_CHM + SZ_VF;
constexpr size_t OFF_RL_RGB = OFF_MX_RGB + SZ_ST;
constexpr size_t OFF_MX_CHM = OFF_RL_RGB + SZ_ST;
constexpr size_t OFF_RL_CHM = OFF_MX_CHM + SZ_ST;
constexpr size_t OFF_O_RGB  = OFF_RL_CHM + SZ_ST;   // attention output fp32 [c][n]
constexpr size_t OFF_O_CHM  = OFF_O_RGB  + SZ_VF;

// ---------------- helpers ----------------
__device__ __forceinline__ unsigned short f2bf(float f) {
  unsigned int u = __float_as_uint(f);
  u += 0x7fffu + ((u >> 16) & 1u);          // round-to-nearest-even
  return (unsigned short)(u >> 16);
}
__device__ __forceinline__ float bf2f(unsigned short h) {
  return __uint_as_float(((unsigned int)h) << 16);
}
__device__ __forceinline__ float fast_exp2(float x) {
  return __builtin_amdgcn_exp2f(x);         // bare v_exp_f32; arg <= 0 here
}
__device__ __forceinline__ v16bf load_bf16_2x8(const unsigned short* p0,
                                               const unsigned short* p1) {
  union { v16bf v; v8bf h[2]; } u;
  u.h[0] = *reinterpret_cast<const v8bf*>(p0);
  u.h[1] = *reinterpret_cast<const v8bf*>(p1);
  return u.v;
}
__device__ __forceinline__ v8f v8f_zero() {
  v8f z;
#pragma unroll
  for (int i = 0; i < 8; ++i) z[i] = 0.0f;
  return z;
}
__device__ __forceinline__ v8f wmma_bf16(v16bf a, v16bf b, v8f c) {
  // D = A(16x32) * B(32x16) + C, fp32 accumulate
  return __builtin_amdgcn_wmma_f32_16x16x32_bf16(false, a, false, b, (short)0, c,
                                                 false, false);
}

// ---------------- K1: bilinear resize of pos_enc 32x32 -> 80x80 ----------------
__global__ void caf_pe_resize(const float* __restrict__ pos, char* ws) {
  int idx = blockIdx.x * 256 + threadIdx.x;
  if (idx >= 32 * HWC) return;
  int o = idx / HWC, p = idx % HWC;
  int y = p / 80, x = p % 80;
  float sy = fminf(fmaxf((y + 0.5f) * 0.4f - 0.5f, 0.0f), 31.0f);
  float sx = fminf(fmaxf((x + 0.5f) * 0.4f - 0.5f, 0.0f), 31.0f);
  int y0 = (int)floorf(sy), x0 = (int)floorf(sx);
  int y1 = min(y0 + 1, 31), x1 = min(x0 + 1, 31);
  float fy = sy - (float)y0, fx = sx - (float)x0;
  const float* s = pos + o * 1024;
  float v = (1.f - fy) * ((1.f - fx) * s[y0 * 32 + x0] + fx * s[y0 * 32 + x1]) +
            fy * ((1.f - fx) * s[y1 * 32 + x0] + fx * s[y1 * 32 + x1]);
  ((float*)(ws + OFF_PE))[idx] = v;
}

// ---------------- K2: q/k/v 1x1-conv projections (+pe for q,k) ----------------
// grid (100, 320, 2), block 64.  rows 0..31=q, 32..63=k, 64..319=v
__global__ __launch_bounds__(64) void caf_qkv_proj(
    const float* __restrict__ rgb, const float* __restrict__ chm,
    const float* __restrict__ q_w, const float* __restrict__ q_b,
    const float* __restrict__ k_w, const float* __restrict__ k_b,
    const float* __restrict__ v_w, const float* __restrict__ v_b, char* ws) {
  const int inst = blockIdx.z;
  const float* x = inst ? chm : rgb;
  const int n = blockIdx.x * 64 + threadIdx.x;
  const int row = blockIdx.y;
  const float* w;
  if (row < 32)       w = q_w + row * 256;
  else if (row < 64)  w = k_w + (row - 32) * 256;
  else                w = v_w + (row - 64) * 256;
  float acc = 0.0f;
#pragma unroll 8
  for (int c = 0; c < 256; ++c) acc = fmaf(w[c], x[c * HWC + n], acc);
  const float* pe = (const float*)(ws + OFF_PE);
  if (row < 32) {
    acc += q_b[row] + pe[row * HWC + n];
    ((unsigned short*)(ws + (inst ? OFF_QT_CHM : OFF_QT_RGB)))[(size_t)n * REDC + row] = f2bf(acc);
  } else if (row < 64) {
    int r = row - 32;
    acc += k_b[r] + pe[r * HWC + n];
    ((unsigned short*)(ws + (inst ? OFF_KT_CHM : OFF_KT_RGB)))[(size_t)n * REDC + r] = f2bf(acc);
  } else {
    int r = row - 64;
    acc += v_b[r];
    ((float*)(ws + (inst ? OFF_VF_CHM : OFF_VF_RGB)))[(size_t)r * HWC + n] = acc;
    ((unsigned short*)(ws + (inst ? OFF_VB_CHM : OFF_VB_RGB)))[(size_t)r * HWC + n] = f2bf(acc);
  }
}

// ---------------- K3: softmax row stats (max, 1/sum) over m, per row n -------
// grid (800, 2), block 256 (8 waves, one row per wave)
__global__ __launch_bounds__(256) void caf_softmax_stats(char* ws) {
  const int inst = blockIdx.y;
  const unsigned short* Qt = (const unsigned short*)(ws + (inst ? OFF_QT_CHM : OFF_QT_RGB));
  const unsigned short* Kt = (const unsigned short*)(ws + (inst ? OFF_KT_RGB : OFF_KT_CHM));
  float* mx2 = (float*)(ws + (inst ? OFF_MX_CHM : OFF_MX_RGB));
  float* rl  = (float*)(ws + (inst ? OFF_RL_CHM : OFF_RL_RGB));
  __shared__ float qs[8][REDC];
  const int wave = threadIdx.x >> 5, lane = threadIdx.x & 31;
  const int n = blockIdx.x * 8 + wave;
  qs[wave][lane] = bf2f(Qt[(size_t)n * REDC + lane]);
  __syncthreads();
  float mx = -__builtin_inff(), sm = 0.0f;
  for (int m = lane; m < HWC; m += 32) {
    const unsigned int* kr = (const unsigned int*)(Kt + (size_t)m * REDC);
    float s = 0.0f;
#pragma unroll
    for (int c = 0; c < 16; ++c) {
      unsigned int w2 = kr[c];
      s = fmaf(qs[wave][2 * c],     __uint_as_float(w2 << 16), s);
      s = fmaf(qs[wave][2 * c + 1], __uint_as_float(w2 & 0xffff0000u), s);
    }
    float xv = s * SCALE_L2E;                 // base-2 domain
    if (xv > mx) { sm *= fast_exp2(mx - xv); mx = xv; }
    sm += fast_exp2(xv - mx);
  }
#pragma unroll
  for (int off = 16; off > 0; off >>= 1) {
    float omx = __shfl_xor(mx, off, 32);
    float osm = __shfl_xor(sm, off, 32);
    float nm = fmaxf(mx, omx);
    sm = sm * fast_exp2(mx - nm) + osm * fast_exp2(omx - nm);
    mx = nm;
  }
  if (lane == 0) { mx2[n] = mx; rl[n] = 1.0f / sm; }
}

// ---------------- K4: streaming attention  O[c,m] = sum_n V[c,n]*P[n,m] ------
// grid (HWC/64, 2 insts, 2 c-splits), block 256 (8 waves).
// Per 64-wide n-tile: 8 waves compute the 16 16x16 S tiles (2 each) via WMMA
// (K-operand hoisted: loop-invariant), exp -> bf16 P (64x64, [m][n]-major) in
// LDS, then each wave runs 8 PV WMMAs accumulating its 16x64 slice of O.
__global__ __launch_bounds__(256) void caf_attention(char* ws) {
  const int inst = blockIdx.y;
  const unsigned short* Qt = (const unsigned short*)(ws + (inst ? OFF_QT_CHM : OFF_QT_RGB));
  const unsigned short* Kt = (const unsigned short*)(ws + (inst ? OFF_KT_RGB : OFF_KT_CHM));
  const unsigned short* Vb = (const unsigned short*)(ws + (inst ? OFF_VB_CHM : OFF_VB_RGB));
  const float* mx2 = (const float*)(ws + (inst ? OFF_MX_CHM : OFF_MX_RGB));
  const float* rl  = (const float*)(ws + (inst ? OFF_RL_CHM : OFF_RL_RGB));
  float* O = (float*)(ws + (inst ? OFF_O_CHM : OFF_O_RGB));

  __shared__ __align__(64) unsigned short Plds[64][64];  // [m_local][n_local] bf16

  const int tid  = threadIdx.x;
  const int wave = tid >> 5;
  const int lane = tid & 31;
  const int hi   = lane >> 4;      // lane half
  const int l16  = lane & 15;
  const int m0   = blockIdx.x * 64;

  const int s_msub = wave >> 1;    // this wave's 16-col quarter of the m tile
  const int s_npar = wave & 1;     // n-subtile parity (handles nsub, nsub+2)
  const int c0     = blockIdx.z * 128 + wave * 16;  // this wave's 16 channels

  // Hoist the S-GEMM K-operand: B[c][m_local] is n-loop invariant.
  const unsigned short* krow =
      Kt + (size_t)(m0 + 16 * s_msub + l16) * REDC + 16 * hi;
  const v16bf b_s = *reinterpret_cast<const v16bf*>(krow);

  v8f acc[4];
#pragma unroll
  for (int ms = 0; ms < 4; ++ms) acc[ms] = v8f_zero();

  for (int n0 = 0; n0 < HWC; n0 += 64) {
    // ---- S tiles: two 16x16 WMMAs per wave, K = RED = 32 ----
    v8f s[2];
#pragma unroll
    for (int t = 0; t < 2; ++t) {
      const int nsub = s_npar + 2 * t;
      const unsigned short* qrow = Qt + (size_t)(n0 + 16 * nsub + l16) * REDC;
      v16bf a_s = load_bf16_2x8(qrow + 16 * hi, qrow + 16 * hi + 8);
      s[t] = wmma_bf16(a_s, b_s, v8f_zero());
    }

    __syncthreads();  // previous iteration's P fully consumed

    // exp + normalize, write bf16 P transposed into LDS
#pragma unroll
    for (int t = 0; t < 2; ++t) {
      const int nsub = s_npar + 2 * t;
      const int nbase = n0 + 16 * nsub + 8 * hi;  // global n of acc element v
#pragma unroll
      for (int v = 0; v < 8; ++v) {
        float p = fast_exp2(fmaf(s[t][v], SCALE_L2E, -mx2[nbase + v])) * rl[nbase + v];
        Plds[16 * s_msub + l16][16 * nsub + 8 * hi + v] = f2bf(p);
      }
    }
    __syncthreads();

    // ---- PV: acc[ms] += V(16x64) * P(64x16), as two K=32 WMMAs ----
#pragma unroll
    for (int kh = 0; kh < 2; ++kh) {
      const unsigned short* vrow = Vb + (size_t)(c0 + l16) * HWC + n0 + 32 * kh;
      v16bf a_v = load_bf16_2x8(vrow + 16 * hi, vrow + 16 * hi + 8);
#pragma unroll
      for (int ms = 0; ms < 4; ++ms) {
        v16bf b_p = *reinterpret_cast<const v16bf*>(
            &Plds[16 * ms + l16][32 * kh + 16 * hi]);
        acc[ms] = wmma_bf16(a_v, b_p, acc[ms]);
      }
    }
  }

  // write O tile: D layout M = v + 8*hi, N = l16
#pragma unroll
  for (int ms = 0; ms < 4; ++ms) {
#pragma unroll
    for (int v = 0; v < 8; ++v) {
      int c = c0 + 8 * hi + v;
      int m = m0 + 16 * ms + l16;
      O[(size_t)c * HWC + m] = acc[ms][v];
    }
  }
}

// ---------------- K5: depthwise 7x7 on V, add into attention output ----------
__global__ __launch_bounds__(64) void caf_dwconv_add(
    const float* __restrict__ rgb_pe_w, const float* __restrict__ rgb_pe_b,
    const float* __restrict__ chm_pe_w, const float* __restrict__ chm_pe_b,
    char* ws) {
  const int inst = blockIdx.z;
  const int n = blockIdx.x * 64 + threadIdx.x;
  const int c = blockIdx.y;
  const float* V = (const float*)(ws + (inst ? OFF_VF_CHM : OFF_VF_RGB)) + (size_t)c * HWC;
  float* O = (float*)(ws + (inst ? OFF_O_CHM : OFF_O_RGB));
  const float* w = (inst ? chm_pe_w : rgb_pe_w) + c * 49;
  float acc = (inst ? chm_pe_b : rgb_pe_b)[c];
  const int y = n / 80, x = n % 80;
#pragma unroll
  for (int dy = 0; dy < 7; ++dy) {
    int yy = y + dy - 3;
    if (yy < 0 || yy >= 80) continue;
#pragma unroll
    for (int dx = 0; dx < 7; ++dx) {
      int xx = x + dx - 3;
      if (xx < 0 || xx >= 80) continue;
      acc = fmaf(w[dy * 7 + dx], V[yy * 80 + xx], acc);
    }
  }
  O[(size_t)c * HWC + n] += acc;
}

// ---------------- K6: projection 1x1 conv (256->256) -> d_out[1],[2] ---------
__global__ __launch_bounds__(64) void caf_proj(
    const float* __restrict__ rgb_proj_w, const float* __restrict__ rgb_proj_b,
    const float* __restrict__ chm_proj_w, const float* __restrict__ chm_proj_b,
    char* ws, float* __restrict__ out) {
  const int inst = blockIdx.z;
  const int n = blockIdx.x * 64 + threadIdx.x;
  const int c = blockIdx.y;
  const float* A = (const float*)(ws + (inst ? OFF_O_CHM : OFF_O_RGB));
  const float* w = (inst ? chm_proj_w : rgb_proj_w) + c * 256;
  float acc = (inst ? chm_proj_b : rgb_proj_b)[c];
#pragma unroll 8
  for (int k = 0; k < 256; ++k) acc = fmaf(w[k], A[(size_t)k * HWC + n], acc);
  out[(size_t)(1 + inst) * CCH * HWC + (size_t)c * HWC + n] = acc;
}

// ---------------- K7: gate (512->256) + sigmoid blend -> d_out[0] ------------
__global__ __launch_bounds__(64) void caf_gate(
    const float* __restrict__ gate_w, const float* __restrict__ gate_b,
    float* __restrict__ out) {
  const int n = blockIdx.x * 64 + threadIdx.x;
  const int c = blockIdx.y;
  const float* rgbA = out + (size_t)CCH * HWC;
  const float* chmA = out + (size_t)2 * CCH * HWC;
  const float* w = gate_w + c * 512;
  float acc = gate_b[c];
#pragma unroll 8
  for (int k = 0; k < 256; ++k) acc = fmaf(w[k], rgbA[(size_t)k * HWC + n], acc);
#pragma unroll 8
  for (int k = 0; k < 256; ++k) acc = fmaf(w[256 + k], chmA[(size_t)k * HWC + n], acc);
  // sigmoid; gate logits are O(1) so bare v_exp_f32 is safe
  float g = 1.0f / (1.0f + fast_exp2(-acc * L2E));
  float r = rgbA[(size_t)c * HWC + n];
  float h = chmA[(size_t)c * HWC + n];
  out[(size_t)c * HWC + n] = r * g + h * (1.0f - g);
}

// ---------------- host launcher ----------------
extern "C" void kernel_launch(void* const* d_in, const int* in_sizes, int n_in,
                              void* d_out, int out_size, void* d_ws, size_t ws_size,
                              hipStream_t stream) {
  (void)in_sizes; (void)n_in; (void)out_size; (void)ws_size;
  const float* rgb        = (const float*)d_in[0];
  const float* chm        = (const float*)d_in[1];
  const float* pos_enc    = (const float*)d_in[2];
  const float* q_w        = (const float*)d_in[3];
  const float* q_b        = (const float*)d_in[4];
  const float* k_w        = (const float*)d_in[5];
  const float* k_b        = (const float*)d_in[6];
  const float* v_w        = (const float*)d_in[7];
  const float* v_b        = (const float*)d_in[8];
  const float* rgb_pe_w   = (const float*)d_in[9];
  const float* rgb_pe_b   = (const float*)d_in[10];
  const float* chm_pe_w   = (const float*)d_in[11];
  const float* chm_pe_b   = (const float*)d_in[12];
  const float* rgb_proj_w = (const float*)d_in[13];
  const float* rgb_proj_b = (const float*)d_in[14];
  const float* chm_proj_w = (const float*)d_in[15];
  const float* chm_proj_b = (const float*)d_in[16];
  const float* gate_w     = (const float*)d_in[17];
  const float* gate_b     = (const float*)d_in[18];
  char* ws = (char*)d_ws;
  float* out = (float*)d_out;

  caf_pe_resize<<<(32 * HWC + 255) / 256, 256, 0, stream>>>(pos_enc, ws);
  caf_qkv_proj<<<dim3(HWC / 64, 320, 2), 64, 0, stream>>>(
      rgb, chm, q_w, q_b, k_w, k_b, v_w, v_b, ws);
  caf_softmax_stats<<<dim3(HWC / 8, 2), 256, 0, stream>>>(ws);
  caf_attention<<<dim3(HWC / 64, 2, 2), 256, 0, stream>>>(ws);
  caf_dwconv_add<<<dim3(HWC / 64, CCH, 2), 64, 0, stream>>>(
      rgb_pe_w, rgb_pe_b, chm_pe_w, chm_pe_b, ws);
  caf_proj<<<dim3(HWC / 64, CCH, 2), 64, 0, stream>>>(
      rgb_proj_w, rgb_proj_b, chm_proj_w, chm_proj_b, ws, out);
  caf_gate<<<dim3(HWC / 64, CCH), 64, 0, stream>>>(gate_w, gate_b, out);
}
